// HigherOrderMultiInputCrossAttention_16587163697409
// MI455X (gfx1250) — matrix operand order repaired and checked by reading, not compiled, not run on hardware
//
#include <hip/hip_runtime.h>
#include <hip/hip_bf16.h>

typedef __attribute__((ext_vector_type(16))) _Float16 v16h;
typedef __attribute__((ext_vector_type(8)))  _Float16 v8h;
typedef __attribute__((ext_vector_type(2)))  _Float16 v2h;
typedef __attribute__((ext_vector_type(8)))  float    v8f;
typedef __attribute__((ext_vector_type(4)))  unsigned tdm_g0_t;
typedef __attribute__((ext_vector_type(8)))  int      tdm_g1_t;
typedef __attribute__((ext_vector_type(4)))  int      tdm_g2_t;

#define B_   2
#define T_   192
#define C_   512
#define H_   8
#define HS_  64
#define ORD_ 3
#define KSPLIT 4
#define KCH  (T_ / KSPLIT)   // 48 k-steps per split
#define NTILE (B_ * H_ * (T_ / 16))  // 192 query tiles

#if __has_builtin(__builtin_amdgcn_tensor_load_to_lds) && __has_builtin(__builtin_amdgcn_s_wait_tensorcnt)
#define USE_TDM 1
#endif

// ---------------- TDM: 1-D contiguous global -> LDS copy (one issue per wave) -------------
// D# per CDNA5 ISA 8.3/8.4: group0 = {flags(count=1), lds_addr, gaddr_lo, gaddr_hi|type=2},
// group1 = {data_size=8B, tensor_dim0, tensor_dim1=1, tile_dim0, strides}. 1-D tile
// (tile_dim1=0 => unused). nbytes must be a multiple of 8 and < 512KB (tile_dim0 is u16).
__device__ __forceinline__ void tdm_copy_1d(const void* gsrc, void* lds_dst, unsigned nbytes) {
#ifdef USE_TDM
  const unsigned long long ga = (unsigned long long)(uintptr_t)gsrc;
  const unsigned lo = (unsigned)(uintptr_t)lds_dst;   // LDS byte offset = low 32b of generic ptr
  const unsigned n8 = nbytes >> 3;                    // 8-byte elements
  tdm_g0_t g0 = { 1u,                                                  // count=1 (valid user D#)
                  lo,                                                  // lds_addr
                  (unsigned)(ga & 0xFFFFFFFFu),                        // global_addr[31:0]
                  (unsigned)((ga >> 32) & 0x01FFFFFFu) | (2u << 30) }; // global_addr[56:32] | type=2
  tdm_g1_t g1 = { (int)(3u << 16),                        // data_size=3 (8 bytes)
                  (int)((n8 & 0xFFFFu) << 16),            // tensor_dim0[15:0]
                  (int)((n8 >> 16) & 0xFFFFu) | (1 << 16),// tensor_dim0[31:16] | tensor_dim1=1
                  (int)((n8 & 0xFFFFu) << 16),            // tile_dim0 = n8
                  0,                                      // tile_dim1=0 (1-D), tile_dim2=0
                  (int)n8,                                // tensor_dim0_stride[31:0]
                  0, 0 };
  tdm_g2_t gz = { 0, 0, 0, 0 };
#if __clang_major__ >= 23
  tdm_g1_t ge = { 0, 0, 0, 0, 0, 0, 0, 0 };
  __builtin_amdgcn_tensor_load_to_lds(g0, g1, gz, gz, ge, 0);
#else
  __builtin_amdgcn_tensor_load_to_lds(g0, g1, gz, gz, 0);
#endif
#endif
}

// ---------------- WMMA operand loaders (wave32; contiguous k = 16*(l>>4)+s, same order for A and B) -------------
// A 16x32 f16: lane holds row l&15; K slots: kb=(l>>4)*16 -> {kb..kb+15}
__device__ __forceinline__ v16h ldA_f16(const _Float16* __restrict__ p, int stride, int lane) {
  const int m = lane & 15, kb = (lane >> 4) * 16;
  return *(const v16h*)(p + (size_t)m * stride + kb);
}

// A-layout broadcast of one row (all 16 M rows identical): for Qk = Q * K2[k,:]
__device__ __forceinline__ v16h ldArow_f16(const _Float16* __restrict__ r0, int lane) {
  const int kb = (lane >> 4) * 16;
  return *(const v16h*)(r0 + kb);
}

// B 32x16 f16 where B[k][n] = w[n0+n][k0+k]; lane holds col l&15, K contiguous 16 per half.
__device__ __forceinline__ v16h ldBt_f16(const _Float16* __restrict__ p, int stride, int lane) {
  const int n = lane & 15, kb = (lane >> 4) * 16;
  return *(const v16h*)(p + (size_t)n * stride + kb);
}

#define WMMA_F16(a, b, c) \
  __builtin_amdgcn_wmma_f32_16x16x32_f16(false, (a), false, (b), (short)0, (c), false, false)

// ---------------- Kernel 0: one-time f32 -> f16 conversion (x, weights) -------------
__global__ __launch_bounds__(256) void cvt_kernel(const float* __restrict__ s,
                                                  _Float16* __restrict__ d, int n2) {
  const float2* s2 = (const float2*)s;
  v2h* d2 = (v2h*)d;
  for (int i = blockIdx.x * 256 + threadIdx.x; i < n2; i += gridDim.x * 256) {
    const float2 v = s2[i];
    v2h o; o[0] = (_Float16)v.x; o[1] = (_Float16)v.y;
    d2[i] = o;
  }
}

// ---------------- Kernel 1: projections  out[t,d] = x[t,:].w[d,:] + bias[d] -------------
// grid = 5*B*H*(T/16) waves; per wave: M=16(t), N=64(d), K=512(c); all-f16 operands.
__global__ __launch_bounds__(32) void proj_kernel(
    const _Float16* __restrict__ x0h, const _Float16* __restrict__ x1h,
    const _Float16* __restrict__ x2h,
    const _Float16* __restrict__ qwh, const _Float16* __restrict__ kwh,
    const _Float16* __restrict__ vwh,
    const float* __restrict__ qb, const float* __restrict__ kb, const float* __restrict__ vb,
    _Float16* __restrict__ Q0, _Float16* __restrict__ K1, _Float16* __restrict__ K2,
    _Float16* __restrict__ V1, _Float16* __restrict__ V2)
{
  const int lane = threadIdx.x;
  int wid = blockIdx.x;
  const int it = wid % (T_ / 16); wid /= (T_ / 16);
  const int h  = wid % H_;        wid /= H_;
  const int b  = wid % B_;        wid /= B_;
  const int p  = wid;             // 0..4: Q0,K1,K2,V1,V2

  const _Float16* x; const _Float16* w; const float* bias; _Float16* out; int ord;
  if      (p == 0) { x = x0h; w = qwh; bias = qb; out = Q0; ord = 0; }
  else if (p == 1) { x = x1h; w = kwh; bias = kb; out = K1; ord = 1; }
  else if (p == 2) { x = x2h; w = kwh; bias = kb; out = K2; ord = 2; }
  else if (p == 3) { x = x1h; w = vwh; bias = vb; out = V1; ord = 1; }
  else             { x = x2h; w = vwh; bias = vb; out = V2; ord = 2; }

  const _Float16* xp = x + ((size_t)b * T_ + it * 16) * C_;
  const _Float16* wp = w + (size_t)(h * ORD_ + ord) * HS_ * C_;
  const float*    bp = bias + (size_t)(h * ORD_ + ord) * HS_;
  _Float16* op = out + (((size_t)b * H_ + h) * T_ + it * 16) * HS_;

  v8f acc[4] = {{}, {}, {}, {}};
  for (int kk = 0; kk < C_; kk += 32) {
    const v16h a = ldA_f16(xp + kk, C_, lane);
#pragma unroll
    for (int nt = 0; nt < 4; ++nt) {
      const v16h bm = ldBt_f16(wp + (size_t)(nt * 16) * C_ + kk, C_, lane);
      acc[nt] = WMMA_F16(a, bm, acc[nt]);
    }
  }
  const int n  = lane & 15;
  const int mh = (lane >> 4) * 8;            // C layout: VGPR r -> row mh+r, col n
#pragma unroll
  for (int nt = 0; nt < 4; ++nt) {
    const float bv = bp[nt * 16 + n];
#pragma unroll
    for (int r = 0; r < 8; ++r)
      op[(size_t)(mh + r) * HS_ + nt * 16 + n] = (_Float16)(acc[nt][r] + bv);
  }
}

// ---------------- Kernel 2: order-3 flash attention core, split-k over 4 chunks -------------
// grid = NTILE*KSPLIT, one wave per block. For k in chunk: Qk = (Q*scale) o K2[k,:];
// S = Qk @ K1^T (16x192, 24 WMMA); online softmax; O += V2[k,:] o (P @ V1) (24 WMMA).
// Writes UN-normalized partial (O, m, l) for the merge kernel.
__global__ __launch_bounds__(32) void attn_kernel(
    const _Float16* __restrict__ Q0g, const _Float16* __restrict__ K1g,
    const _Float16* __restrict__ K2g, const _Float16* __restrict__ V1g,
    const _Float16* __restrict__ V2g,
    float* __restrict__ Pm, float* __restrict__ Pl, float* __restrict__ PO)
{
  __shared__ __attribute__((aligned(32))) _Float16 sK1[T_ * HS_];   // [j][d]
  __shared__ __attribute__((aligned(32))) _Float16 sV1t[HS_ * T_];  // [d][j]
  __shared__ __attribute__((aligned(32))) _Float16 sP[16 * T_];     // [m][j] staging

  const int lane = threadIdx.x;
  int wid = blockIdx.x;
  const int kc = wid % KSPLIT;    wid /= KSPLIT;
  const int it = wid % (T_ / 16); wid /= (T_ / 16);
  const int h  = wid % H_;        wid /= H_;
  const int b  = wid;
  const int tile = ((b * H_) + h) * (T_ / 16) + it;

  const size_t bh = ((size_t)b * H_ + h) * T_ * HS_;
  const _Float16* K1p = K1g + bh;
  const _Float16* V1p = V1g + bh;
  const _Float16* K2p = K2g + bh;
  const _Float16* V2p = V2g + bh;
  const _Float16* Qp  = Q0g + bh + (size_t)it * 16 * HS_;

  // K1 panel via Tensor Data Mover (falls back to a cooperative copy if no builtin).
#ifdef USE_TDM
  tdm_copy_1d(K1p, sK1, T_ * HS_ * sizeof(_Float16));
#else
  {
    const uint4* src = (const uint4*)K1p;
    uint4* dst = (uint4*)sK1;
    for (int i = lane; i < T_ * HS_ / 8; i += 32) dst[i] = src[i];
  }
#endif
  // V1 transpose fill: [j][d] -> [d][j]
  for (int i = lane; i < T_ * HS_; i += 32) {
    const int j = i / HS_, d = i % HS_;
    sV1t[d * T_ + j] = V1p[i];
  }
#ifdef USE_TDM
  __builtin_amdgcn_s_wait_tensorcnt(0);
#endif
  __syncthreads();

  // Q tile in A layout (two K=32 slabs over HS=64), pre-scaled by 1/sqrt(HS)=2^-3 (exact).
  v16h q[2];
#pragma unroll
  for (int s = 0; s < 2; ++s) q[s] = ldA_f16(Qp + s * 32, HS_, lane) * (_Float16)0.125f;

  const int n  = lane & 15;
  const int mh = (lane >> 4) * 8;

  float mst[8], lst[8];
  v8f oacc[4] = {{}, {}, {}, {}};
#pragma unroll
  for (int r = 0; r < 8; ++r) { mst[r] = -1e30f; lst[r] = 0.f; }

  const int kq0 = kc * KCH;
  for (int kq = kq0; kq < kq0 + KCH; ++kq) {
    __builtin_prefetch(K2p + (size_t)(kq + 1) * HS_, 0, 0);   // -> global_prefetch_b8
    __builtin_prefetch(V2p + (size_t)(kq + 1) * HS_, 0, 0);

    // Qk = Qs o K2[kq,:] in A layout (v_pk_mul_f16)
    v16h qk[2];
#pragma unroll
    for (int s = 0; s < 2; ++s) qk[s] = q[s] * ldArow_f16(K2p + (size_t)kq * HS_ + s * 32, lane);

    // S = Qk @ K1^T : 12 N-tiles x 2 K-steps
    v8f sacc[12];
#pragma unroll
    for (int jn = 0; jn < 12; ++jn) {
      v8f acc = {};
#pragma unroll
      for (int s = 0; s < 2; ++s) {
        const int kb2 = (lane >> 4) * 16;
        const v16h bm = *(const v16h*)&sK1[(size_t)(jn * 16 + n) * HS_ + s * 32 + kb2];
        acc = WMMA_F16(qk[s], bm, acc);
      }
      sacc[jn] = acc;
    }

    // Row max of this slab (reduce across the 16-lane half holding each row).
    float mrow[8];
#pragma unroll
    for (int r = 0; r < 8; ++r) mrow[r] = -1e30f;
#pragma unroll
    for (int jn = 0; jn < 12; ++jn)
#pragma unroll
      for (int r = 0; r < 8; ++r) mrow[r] = fmaxf(mrow[r], sacc[jn][r]);
#pragma unroll
    for (int off = 1; off < 16; off <<= 1)
#pragma unroll
      for (int r = 0; r < 8; ++r) mrow[r] = fmaxf(mrow[r], __shfl_xor(mrow[r], off, 32));

    float corr[8], rs[8];
#pragma unroll
    for (int r = 0; r < 8; ++r) {
      const float mn = fmaxf(mst[r], mrow[r]);
      corr[r] = __expf(mst[r] - mn);
      mst[r]  = mn;
      rs[r]   = 0.f;
    }

    // P = exp(S - m): accumulate row sums, stage [m][j] to LDS for the A-layout re-read.
#pragma unroll
    for (int jn = 0; jn < 12; ++jn)
#pragma unroll
      for (int r = 0; r < 8; ++r) {
        const float pe = __expf(sacc[jn][r] - mst[r]);
        rs[r] += pe;
        sP[(size_t)(mh + r) * T_ + jn * 16 + n] = (_Float16)pe;
      }
#pragma unroll
    for (int off = 1; off < 16; off <<= 1)
#pragma unroll
      for (int r = 0; r < 8; ++r) rs[r] += __shfl_xor(rs[r], off, 32);
#pragma unroll
    for (int r = 0; r < 8; ++r) lst[r] = lst[r] * corr[r] + rs[r];
#pragma unroll
    for (int nt = 0; nt < 4; ++nt)
#pragma unroll
      for (int r = 0; r < 8; ++r) oacc[nt][r] *= corr[r];

    __syncthreads();   // single wave: DS ops are in-order; keep for compiler ordering

    // PV = P @ V1 : 4 N-tiles x 6 K-steps; then scale column d by V2[kq,d].
#pragma unroll
    for (int nt = 0; nt < 4; ++nt) {
      v8f pv = {};
#pragma unroll
      for (int s = 0; s < 6; ++s) {
        const int m  = lane & 15;
        const int kb = (lane >> 4) * 16;
        const v16h a = *(const v16h*)&sP[(size_t)m * T_ + s * 32 + kb];
        const int kb2 = (lane >> 4) * 16;
        const v16h bm = *(const v16h*)&sV1t[(size_t)(nt * 16 + n) * T_ + s * 32 + kb2];
        pv = WMMA_F16(a, bm, pv);
      }
      const float v2v = (float)V2p[(size_t)kq * HS_ + nt * 16 + n];
#pragma unroll
      for (int r = 0; r < 8; ++r) oacc[nt][r] += pv[r] * v2v;
    }
  }

  // Write un-normalized partial accumulator + per-row (m, l) state.
  float* POp = PO + (size_t)(tile * KSPLIT + kc) * 16 * HS_;
#pragma unroll
  for (int nt = 0; nt < 4; ++nt)
#pragma unroll
    for (int r = 0; r < 8; ++r)
      POp[(size_t)(mh + r) * HS_ + nt * 16 + n] = oacc[nt][r];
  if (n == 0) {
#pragma unroll
    for (int r = 0; r < 8; ++r) {
      Pm[(tile * KSPLIT + kc) * 16 + mh + r] = mst[r];
      Pl[(tile * KSPLIT + kc) * 16 + mh + r] = lst[r];
    }
  }
}

// ---------------- Kernel 3: log-sum-exp merge of the 4 split-k partials -> Yh (f16) ------
__global__ __launch_bounds__(32) void merge_kernel(
    const float* __restrict__ Pm, const float* __restrict__ Pl, const float* __restrict__ PO,
    _Float16* __restrict__ Yh)
{
  const int tile = blockIdx.x;            // ((b*H + h)*12 + it)
  const int it = tile % (T_ / 16);
  const int bh = tile / (T_ / 16);
  const int h  = bh % H_;
  const int b  = bh / H_;
  for (int idx = threadIdx.x; idx < 16 * HS_; idx += 32) {
    const int row = idx >> 6, col = idx & (HS_ - 1);
    float ms = -1e30f;
#pragma unroll
    for (int w = 0; w < KSPLIT; ++w)
      ms = fmaxf(ms, Pm[(tile * KSPLIT + w) * 16 + row]);
    float acc = 0.f, l = 0.f;
#pragma unroll
    for (int w = 0; w < KSPLIT; ++w) {
      const float f = __expf(Pm[(tile * KSPLIT + w) * 16 + row] - ms);
      acc += f * PO[((size_t)(tile * KSPLIT + w) * 16 + row) * HS_ + col];
      l   += f * Pl[(tile * KSPLIT + w) * 16 + row];
    }
    const int t = it * 16 + row;
    Yh[((size_t)b * T_ + t) * C_ + h * HS_ + col] = (_Float16)(acc / l);
  }
}

// ---------------- Kernel 4: output projection  out = Y @ cw^T + cb (f32 out) -------------
__global__ __launch_bounds__(32) void outproj_kernel(
    const _Float16* __restrict__ Yh, const _Float16* __restrict__ cwh,
    const float* __restrict__ cb, float* __restrict__ out)
{
  const int lane = threadIdx.x;
  int wid = blockIdx.x;
  const int nt4 = wid % (C_ / 64); wid /= (C_ / 64);
  const int mt  = wid;

  const _Float16* a0 = Yh + (size_t)mt * 16 * C_;
  v8f acc[4] = {{}, {}, {}, {}};
  for (int kk = 0; kk < C_; kk += 32) {
    const v16h a = ldA_f16(a0 + kk, C_, lane);
#pragma unroll
    for (int nt = 0; nt < 4; ++nt) {
      const v16h bm = ldBt_f16(cwh + (size_t)(nt4 * 64 + nt * 16) * C_ + kk, C_, lane);
      acc[nt] = WMMA_F16(a, bm, acc[nt]);
    }
  }
  const int n  = lane & 15;
  const int mh = (lane >> 4) * 8;
#pragma unroll
  for (int nt = 0; nt < 4; ++nt) {
    const float bv = cb[nt4 * 64 + nt * 16 + n];
#pragma unroll
    for (int r = 0; r < 8; ++r)
      out[(size_t)(mt * 16 + mh + r) * C_ + nt4 * 64 + nt * 16 + n] = acc[nt][r] + bv;
  }
}

// ---------------- Launcher ----------------
extern "C" void kernel_launch(void* const* d_in, const int* in_sizes, int n_in,
                              void* d_out, int out_size, void* d_ws, size_t ws_size,
                              hipStream_t stream) {
  const float* x0 = (const float*)d_in[0];
  const float* x1 = (const float*)d_in[1];
  const float* x2 = (const float*)d_in[2];
  const float* qw = (const float*)d_in[3];
  const float* qb = (const float*)d_in[4];
  const float* kw = (const float*)d_in[5];
  const float* kb = (const float*)d_in[6];
  const float* vw = (const float*)d_in[7];
  const float* vb = (const float*)d_in[8];
  const float* cw = (const float*)d_in[9];
  const float* cb = (const float*)d_in[10];

  const size_t BTC = (size_t)B_ * T_ * C_;             // 196608
  const size_t WSZ = (size_t)H_ * ORD_ * HS_ * C_;     // 786432
  const size_t CWZ = (size_t)C_ * C_;                  // 262144
  const size_t PER = (size_t)B_ * H_ * T_ * HS_;       // 196608

  _Float16* f16p = (_Float16*)d_ws;
  _Float16* x0h = f16p;  f16p += BTC;
  _Float16* x1h = f16p;  f16p += BTC;
  _Float16* x2h = f16p;  f16p += BTC;
  _Float16* qwh = f16p;  f16p += WSZ;
  _Float16* kwh = f16p;  f16p += WSZ;
  _Float16* vwh = f16p;  f16p += WSZ;
  _Float16* cwh = f16p;  f16p += CWZ;
  _Float16* Q0  = f16p;  f16p += PER;
  _Float16* K1  = f16p;  f16p += PER;
  _Float16* K2  = f16p;  f16p += PER;
  _Float16* V1  = f16p;  f16p += PER;
  _Float16* V2  = f16p;  f16p += PER;
  _Float16* Yh  = f16p;  f16p += BTC;
  float* fp = (float*)f16p;                            // f16 bytes so far are 4B-aligned
  float* Pm = fp;  fp += (size_t)NTILE * KSPLIT * 16;
  float* Pl = fp;  fp += (size_t)NTILE * KSPLIT * 16;
  float* PO = fp;  fp += (size_t)NTILE * KSPLIT * 16 * HS_;

  auto cvt = [&](const float* s, _Float16* d, size_t n) {
    const int n2 = (int)(n / 2);
    int blocks = (n2 + 255) / 256; if (blocks > 1024) blocks = 1024;
    cvt_kernel<<<blocks, 256, 0, stream>>>(s, d, n2);
  };
  cvt(x0, x0h, BTC); cvt(x1, x1h, BTC); cvt(x2, x2h, BTC);
  cvt(qw, qwh, WSZ); cvt(kw, kwh, WSZ); cvt(vw, vwh, WSZ);
  cvt(cw, cwh, CWZ);

  proj_kernel<<<5 * B_ * H_ * (T_ / 16), 32, 0, stream>>>(
      x0h, x1h, x2h, qwh, kwh, vwh, qb, kb, vb, Q0, K1, K2, V1, V2);

  attn_kernel<<<NTILE * KSPLIT, 32, 0, stream>>>(Q0, K1, K2, V1, V2, Pm, Pl, PO);

  merge_kernel<<<NTILE, 32, 0, stream>>>(Pm, Pl, PO, Yh);

  outproj_kernel<<<(B_ * T_ / 16) * (C_ / 64), 32, 0, stream>>>(Yh, cwh, cb, (float*)d_out);
}
